// GNNEncoder_6193342841617
// MI455X (gfx1250) — matrix operand order repaired and checked by reading, not compiled, not run on hardware
//
#include <hip/hip_runtime.h>
#include <hip/hip_bf16.h>
#include <math.h>

// ---- problem constants (match reference) ----
#define HC     80   // heads * channels
#define HEADS  5
#define CH     16
#define EDIM   16
#define NEG_SLOPE 0.2f

typedef __attribute__((ext_vector_type(16))) _Float16 v16h;
typedef __attribute__((ext_vector_type(8)))  float    v8f;

// sign-aware float atomic max (memory pre-initialized to -inf bits)
__device__ __forceinline__ void atomicMaxFloat(float* addr, float val) {
    if (val >= 0.0f) atomicMax((int*)addr, __float_as_int(val));
    else             atomicMin((unsigned int*)addr, (unsigned int)__float_as_int(val));
}

// =====================================================================
// WMMA GEMM: OUT[m, 0..79] = X[m, :] * W^T + bias,  W: (80, K) row-major.
// 128 rows / workgroup, 8 waves, each wave owns a 16x80 strip.
// LDS tiles are staged ALREADY IN WMMA FRAGMENT ORDER:
//   sA[wave][ks][lane][slot0..15]  (32B contiguous per lane -> ds_load_b128 x2)
//   sB[ks][t][lane][slot0..15]
// A mapping (16-bit A 16x32): lane=(r&15)+((k&16)?16:0),
//   slot=(k&15)  -> an 8-aligned k-octet is one b128 store.
// B mapping (16-bit B 32x16): lane: col=t*16+(lane&15), k=ks*32+((lane&16)?16:0)+slot
//   -> 16 consecutive k of one W row = 64B contiguous global read.
// =====================================================================
template <int K, int KPAD>
__global__ __launch_bounds__(256)
void gemm_xw_bias_wmma(const float* __restrict__ X, const float* __restrict__ W,
                       const float* __restrict__ bias, float* __restrict__ OUT,
                       int M) {
    constexpr int NKS = KPAD / 32;              // k-steps of 32
    constexpr int OCT = KPAD / 8;               // 8-element k-octets per row
    __shared__ __align__(32) _Float16 sA[8 * NKS * 32 * 16];
    __shared__ __align__(32) _Float16 sB[NKS * 5 * 32 * 16];

    const int tid   = threadIdx.x;
    const int mbase = blockIdx.x * 128;

    // ---- stage A (128 x KPAD) in fragment order, f32->f16, zero-padded ----
    for (int idx = tid; idx < 128 * OCT; idx += 256) {
        const int r   = idx / OCT;
        const int k0  = (idx - r * OCT) * 8;
        const int gr  = mbase + r;
        float v[8];
        if (gr < M && k0 < K) {                       // K multiple of 8: full octet
            const float4* p = (const float4*)(X + (size_t)gr * K + k0);
            *(float4*)&v[0] = p[0]; *(float4*)&v[4] = p[1];
        } else {
#pragma unroll
            for (int i = 0; i < 8; ++i) v[i] = 0.0f;
        }
        __align__(16) _Float16 h[8];
#pragma unroll
        for (int i = 0; i < 8; ++i) h[i] = (_Float16)v[i];
        const int wave = r >> 4, rlo = r & 15;
        const int ks = k0 >> 5, kin = k0 & 31;
        const int lane = rlo + ((kin & 16) ? 16 : 0);
        const int slot = kin & 8;
        *(uint4*)&sA[((wave * NKS + ks) * 32 + lane) * 16 + slot] = *(const uint4*)h;
    }

    // ---- stage B = W^T (KPAD x 80) in fragment order ----
    for (int idx = tid; idx < NKS * 5 * 32; idx += 256) {
        const int ks   = idx / (5 * 32);
        const int rem  = idx - ks * 5 * 32;
        const int t    = rem >> 5;
        const int lane = rem & 31;
        const int col   = t * 16 + (lane & 15);
        const int kbase = ks * 32 + ((lane & 16) ? 16 : 0);
        float v[16];
#pragma unroll
        for (int i = 0; i < 16; i += 4) {
            if (kbase + i < K) {                      // K multiple of 16: full quads
                *(float4*)&v[i] = *(const float4*)(W + (size_t)col * K + kbase + i);
            } else {
                v[i] = v[i+1] = v[i+2] = v[i+3] = 0.0f;
            }
        }
        __align__(16) _Float16 h[16];
#pragma unroll
        for (int i = 0; i < 16; ++i) h[i] = (_Float16)v[i];
        _Float16* dst = &sB[((ks * 5 + t) * 32 + lane) * 16];
        ((uint4*)dst)[0] = ((const uint4*)h)[0];
        ((uint4*)dst)[1] = ((const uint4*)h)[1];
    }
    __syncthreads();

    const int wave = tid >> 5, lane = tid & 31;
    v8f acc[5];
#pragma unroll
    for (int t = 0; t < 5; ++t) acc[t] = (v8f){0,0,0,0,0,0,0,0};

    const v16h* aP = (const v16h*)&sA[(wave * NKS) * 32 * 16];
    const v16h* bP = (const v16h*)&sB[0];
#pragma unroll
    for (int ks = 0; ks < NKS; ++ks) {
        const v16h af = aP[ks * 32 + lane];
#pragma unroll
        for (int t = 0; t < 5; ++t) {
            const v16h bf = bP[(ks * 5 + t) * 32 + lane];
            acc[t] = __builtin_amdgcn_wmma_f32_16x16x32_f16(
                false, af, false, bf, (short)0, acc[t], false, false);
        }
    }

    // C/D layout: VGPR r -> (M = wave*16 + r (+8 for lanes 16-31), N = lane&15)
    const int rbase = mbase + wave * 16 + ((lane & 16) ? 8 : 0);
#pragma unroll
    for (int t = 0; t < 5; ++t) {
        const int col = t * 16 + (lane & 15);
        const float b = bias[col];
#pragma unroll
        for (int r = 0; r < 8; ++r) {
            const int grow = rbase + r;
            if (grow < M) OUT[(size_t)grow * HC + col] = acc[t][r] + b;
        }
    }
}

// =====================================================================
// Edge pass 1: alpha[h][e] = sum_c att[h,c]*lrelu(xl[src]+xr[dst]+em)
// em recomputed in-register from edge_attr against LDS-resident we slice
// (avoids materializing the 320MB E x HC tensor). atomic-max into amax.
// =====================================================================
__global__ __launch_bounds__(256)
void edge_alpha_pass(const int* __restrict__ ei, const float* __restrict__ eattr,
                     const float* __restrict__ XL, const float* __restrict__ XR,
                     const float* __restrict__ we, const float* __restrict__ att,
                     float* __restrict__ ALPHA, float* __restrict__ AMAX,
                     int E, int Nn) {
    __shared__ float sWe[CH * EDIM];   // head-h slice of we: [c][d]
    __shared__ float sAtt[CH];
    const int h   = blockIdx.y;
    const int tid = threadIdx.x;
    if (tid < CH * EDIM) sWe[tid] = we[(size_t)(h * CH + tid / EDIM) * EDIM + (tid % EDIM)];
    if (tid < CH)        sAtt[tid] = att[h * CH + tid];
    __syncthreads();

    const int e = blockIdx.x * 256 + tid;
    if (e >= E) return;
    const int src = ei[e], dst = ei[E + e];

    float ea[EDIM], xs[CH], xd[CH];
    {
        const float4* p = (const float4*)(eattr + (size_t)e * EDIM);
        *(float4*)&ea[0] = p[0]; *(float4*)&ea[4] = p[1];
        *(float4*)&ea[8] = p[2]; *(float4*)&ea[12] = p[3];
        const float4* l = (const float4*)(XL + (size_t)src * HC + h * CH);
        *(float4*)&xs[0] = l[0]; *(float4*)&xs[4] = l[1];
        *(float4*)&xs[8] = l[2]; *(float4*)&xs[12] = l[3];
        const float4* r = (const float4*)(XR + (size_t)dst * HC + h * CH);
        *(float4*)&xd[0] = r[0]; *(float4*)&xd[4] = r[1];
        *(float4*)&xd[8] = r[2]; *(float4*)&xd[12] = r[3];
    }
    float alpha = 0.0f;
#pragma unroll
    for (int c = 0; c < CH; ++c) {
        float em = 0.0f;
#pragma unroll
        for (int d = 0; d < EDIM; ++d) em = fmaf(ea[d], sWe[c * EDIM + d], em);
        float m = xs[c] + xd[c] + em;
        m = (m > 0.0f) ? m : m * NEG_SLOPE;
        alpha = fmaf(sAtt[c], m, alpha);
    }
    ALPHA[(size_t)h * E + e] = alpha;
    atomicMaxFloat(&AMAX[(size_t)h * Nn + dst], alpha);
}

// Edge pass 2: ex = exp(alpha - amax[dst]); den[dst] += ex (segment softmax denom)
__global__ __launch_bounds__(256)
void edge_softmax_pass(const int* __restrict__ ei, float* __restrict__ ALPHA,
                       const float* __restrict__ AMAX, float* __restrict__ DEN,
                       int E, int Nn) {
    const int h = blockIdx.y;
    const int e = blockIdx.x * 256 + threadIdx.x;
    if (e >= E) return;
    const int dst = ei[E + e];
    float ex = __expf(ALPHA[(size_t)h * E + e] - AMAX[(size_t)h * Nn + dst]);
    ALPHA[(size_t)h * E + e] = ex;
    atomicAdd(&DEN[(size_t)h * Nn + dst], ex);
}

// Edge pass 3: out[dst] += xl[src] * (ex / den[dst])
__global__ __launch_bounds__(256)
void edge_scatter_pass(const int* __restrict__ ei, const float* __restrict__ ALPHA,
                       const float* __restrict__ DEN, const float* __restrict__ XL,
                       float* __restrict__ ACC, int E, int Nn) {
    const int h = blockIdx.y;
    const int e = blockIdx.x * 256 + threadIdx.x;
    if (e >= E) return;
    const int src = ei[e], dst = ei[E + e];
    const float a = ALPHA[(size_t)h * E + e] / (DEN[(size_t)h * Nn + dst] + 1e-16f);
    const float* xs = XL + (size_t)src * HC + h * CH;
    float* op = ACC + (size_t)dst * HC + h * CH;
#pragma unroll
    for (int c = 0; c < CH; ++c) atomicAdd(&op[c], xs[c] * a);
}

// per-layer scratch init: acc=0, amax=-inf, den=0
__global__ __launch_bounds__(256)
void layer_init(float* __restrict__ ACC, float* __restrict__ AMAX,
                float* __restrict__ DEN, int Nn) {
    const int i = blockIdx.x * 256 + threadIdx.x;
    if (i < Nn * HC) ACC[i] = 0.0f;
    if (i < Nn * HEADS) { AMAX[i] = __int_as_float(0xFF800000); DEN[i] = 0.0f; }
}

// finalize: out = acc + bias (ReLU for layers 1-2)
__global__ __launch_bounds__(256)
void node_finalize(const float* __restrict__ ACC, const float* __restrict__ bias,
                   float* __restrict__ OUT, int Nn, int relu) {
    const int i = blockIdx.x * 256 + threadIdx.x;
    if (i >= Nn * HC) return;
    float v = ACC[i] + bias[i % HC];
    if (relu) v = fmaxf(v, 0.0f);
    OUT[i] = v;
}

extern "C" void kernel_launch(void* const* d_in, const int* in_sizes, int n_in,
                              void* d_out, int out_size, void* d_ws, size_t ws_size,
                              hipStream_t stream) {
    const float* x     = (const float*)d_in[0];
    const int*   ei    = (const int*)d_in[1];
    const float* eattr = (const float*)d_in[2];
    const int N = in_sizes[0] / 128;   // F = 128
    const int E = in_sizes[1] / 2;

    const float *wl[3], *bl[3], *wr[3], *br[3], *we[3], *att[3], *bias[3];
    for (int L = 0; L < 3; ++L) {
        int b = 3 + L * 7;
        wl[L]   = (const float*)d_in[b + 0];
        bl[L]   = (const float*)d_in[b + 1];
        wr[L]   = (const float*)d_in[b + 2];
        br[L]   = (const float*)d_in[b + 3];
        we[L]   = (const float*)d_in[b + 4];
        att[L]  = (const float*)d_in[b + 5];
        bias[L] = (const float*)d_in[b + 6];
    }

    // workspace carve (~152 MB)
    float* ws = (float*)d_ws;
    const size_t nHC = (size_t)N * HC;
    float* XL    = ws;                         // N*HC
    float* XR    = XL + nHC;                   // N*HC
    float* Hb    = XR + nHC;                   // N*HC  (layer output, next input)
    float* ACC   = Hb + nHC;                   // N*HC  (pre-bias accumulator)
    float* ALPHA = ACC + nHC;                  // HEADS*E
    float* AMAX  = ALPHA + (size_t)HEADS * E;  // HEADS*N
    float* DEN   = AMAX + (size_t)HEADS * N;   // HEADS*N

    dim3 blk(256);
    dim3 gGemm((N + 127) / 128);
    dim3 gEdge((E + 255) / 256, HEADS);
    dim3 gNode((N * HC + 255) / 256);

    const float* cur = x;
    for (int L = 0; L < 3; ++L) {
        layer_init<<<gNode, blk, 0, stream>>>(ACC, AMAX, DEN, N);
        if (L == 0) {
            gemm_xw_bias_wmma<128, 128><<<gGemm, blk, 0, stream>>>(cur, wl[L], bl[L], XL, N);
            gemm_xw_bias_wmma<128, 128><<<gGemm, blk, 0, stream>>>(cur, wr[L], br[L], XR, N);
        } else {
            gemm_xw_bias_wmma<80, 96><<<gGemm, blk, 0, stream>>>(cur, wl[L], bl[L], XL, N);
            gemm_xw_bias_wmma<80, 96><<<gGemm, blk, 0, stream>>>(cur, wr[L], br[L], XR, N);
        }
        edge_alpha_pass<<<gEdge, blk, 0, stream>>>(ei, eattr, XL, XR, we[L], att[L],
                                                   ALPHA, AMAX, E, N);
        edge_softmax_pass<<<gEdge, blk, 0, stream>>>(ei, ALPHA, AMAX, DEN, E, N);
        edge_scatter_pass<<<gEdge, blk, 0, stream>>>(ei, ALPHA, DEN, XL, ACC, E, N);
        float* tgt = (L == 2) ? (float*)d_out : Hb;
        node_finalize<<<gNode, blk, 0, stream>>>(ACC, bias[L], tgt, N, (L < 2) ? 1 : 0);
        cur = Hb;
    }
}
